// MultiHeadedEdgeAttention_37881611551379
// MI455X (gfx1250) — matrix operand order repaired and checked by reading, not compiled, not run on hardware
//
#include <hip/hip_runtime.h>
#include <hip/hip_bf16.h>
#include <stdint.h>

typedef __attribute__((ext_vector_type(16))) _Float16 v16h;
typedef __attribute__((ext_vector_type(8)))  float    v8f;
typedef __attribute__((ext_vector_type(4)))  unsigned int u32x4;
typedef __attribute__((ext_vector_type(2)))  unsigned int u32x2;

#define EFW 1024   // concat width: query|edge|reverse_edge|value
#define HID 768    // nn_edge hidden width

union FragU { v16h h; u32x4 q[2]; };

// A-matrix 16x32 f16 fragment (row-major source, stride in halves).
// Lanes 0-15: row=lane, K in {0..7, 16..23}; lanes 16-31: row=lane-16, K in {8..15, 24..31}.
__device__ __forceinline__ v16h load_a_frag(const _Float16* base, int stride, int kbase, int lane) {
    const int row = lane & 15;
    const int khi = (lane >> 4) << 3;                    // 0 or 8 halves
    const u32x4* p = (const u32x4*)(base + (size_t)row * stride + kbase + khi);
    FragU f; f.q[0] = p[0]; f.q[1] = p[2];               // second group at +16 halves
    return f.h;
}

// B-matrix 32x16 f16 fragment from n-major weights (stride = K in halves).
// Lanes 0-15: col=lane, K in {0..7, 16..23}; lanes 16-31: col=lane-16, K in {8..15, 24..31}.
__device__ __forceinline__ v16h load_b_frag(const _Float16* base, int stride, int kbase, int lane) {
    const int col = lane & 15;
    const int khi = (lane >> 4) << 3;                    // 0 or 8 halves
    const u32x4* p = (const u32x4*)(base + (size_t)col * stride + kbase + khi);
    FragU f; f.q[0] = p[0]; f.q[1] = p[2];
    return f.h;
}

__device__ __forceinline__ v8f wmma16(v16h a, v16h b, v8f c) {
    return __builtin_amdgcn_wmma_f32_16x16x32_f16(false, a, false, b, (short)0, c, false, false);
}

// Transpose-convert f32 [K,N] row-major -> f16 n-major [N][K]
__global__ void cvtT_kernel(const float* __restrict__ W, _Float16* __restrict__ out, int K, int Nn) {
    int t = blockIdx.x * blockDim.x + threadIdx.x;
    if (t >= K * Nn) return;
    int k = t / Nn, n = t % Nn;
    out[(size_t)n * K + k] = (_Float16)W[t];
}

__global__ void __launch_bounds__(32)
mhea_kernel(const float* __restrict__ query, const float* __restrict__ edge,
            const float* __restrict__ value, const float* __restrict__ redge,
            const float* __restrict__ srcc,  const float* __restrict__ dstc,
            const float* __restrict__ be1,   const float* __restrict__ be2,
            const float* __restrict__ bq,    const float* __restrict__ bpe,
            const float* __restrict__ bvv,   const float* __restrict__ a1b,
            const float* __restrict__ a2b,
            const _Float16* __restrict__ wB1,  const _Float16* __restrict__ wB2,
            const _Float16* __restrict__ wWq,  const _Float16* __restrict__ wWpe,
            const _Float16* __restrict__ wWv,  const _Float16* __restrict__ wA1,
            const _Float16* __restrict__ wA2,
            float* __restrict__ outX, float* __restrict__ outEF,
            float* __restrict__ outP, int Nedges)
{
    __shared__ __align__(16) _Float16 efL[16 * EFW];   // 32 KB staged activations (f16)
    __shared__ __align__(16) _Float16 xL[128 * 64];    // 16 KB attention input X
    __shared__ __align__(16) _Float16 scL[16 * 64];    //  2 KB bounce tile
    __shared__ float dwL[16];
    __shared__ float sL[16];

    const int lane  = threadIdx.x;            // 0..31 (wave32)
    const int base  = blockIdx.x * 16;
    const int ncol  = lane & 15;              // C-matrix: column
    const int mbase = (lane >> 4) << 3;       // C-matrix: row base (0 or 8)

    // ---------------- stage ef_in = [q|e|re|v] as f16 into LDS ----------------
    const float* srcs[4] = { query, edge, redge, value };
    for (int it = 0; it < 128; ++it) {
        int g   = (it * 32 + lane) * 4;       // flat half index in 16x1024
        int row = g >> 10;
        int col = g & 1023;
        int e = base + row; if (e >= Nedges) e = Nedges - 1;
        const float4* fp = (const float4*)(srcs[col >> 8] + (size_t)e * 256 + (col & 255));
        float4 v = *fp;
        union { _Float16 h[4]; u32x2 u; } pk;
        pk.h[0] = (_Float16)v.x; pk.h[1] = (_Float16)v.y;
        pk.h[2] = (_Float16)v.z; pk.h[3] = (_Float16)v.w;
        *(u32x2*)(efL + row * EFW + col) = pk.u;
    }
    if (lane < 16) {
        int e = base + lane; if (e >= Nedges) e = Nedges - 1;
        float dx = srcc[e*3+0] - dstc[e*3+0];
        float dy = srcc[e*3+1] - dstc[e*3+1];
        float dz = srcc[e*3+2] - dstc[e*3+2];
        float d2 = dx*dx + dy*dy + dz*dz;
        float dw = __expf(-0.5f * d2);        // SIGMA = 1
        dwL[lane] = dw;
        sL[lane]  = dw / (dw + 1e-10f);       // sum_c softmax == 1 -> final scale
    }
    __syncthreads();

    // ------------- GEMM1 (ef@We1, relu) fused with GEMM2 (@We2) --------------
    v8f C2[16];
#pragma unroll
    for (int nt = 0; nt < 16; ++nt) C2[nt] = (v8f){};

#pragma unroll 1
    for (int kt2 = 0; kt2 < 24; ++kt2) {      // hidden cols in blocks of 32
        int hb = kt2 * 32;
        v8f acc0 = (v8f){}, acc1 = (v8f){};
        const _Float16* b0base = wB1 + (size_t)hb * EFW;
        const _Float16* b1base = wB1 + (size_t)(hb + 16) * EFW;
#pragma unroll 4
        for (int k1 = 0; k1 < 32; ++k1) {     // K = 1024
            v16h a  = load_a_frag(efL, EFW, k1 * 32, lane);
            v16h b0 = load_b_frag(b0base, EFW, k1 * 32, lane);
            v16h b1 = load_b_frag(b1base, EFW, k1 * 32, lane);
            acc0 = wmma16(a, b0, acc0);
            acc1 = wmma16(a, b1, acc1);
        }
        float bb0 = be1[hb + ncol], bb1 = be1[hb + 16 + ncol];
#pragma unroll
        for (int v = 0; v < 8; ++v) {
            float t0 = acc0[v] + bb0; t0 = t0 > 0.f ? t0 : 0.f;
            float t1 = acc1[v] + bb1; t1 = t1 > 0.f ? t1 : 0.f;
            scL[(mbase + v) * 32 + ncol]      = (_Float16)t0;
            scL[(mbase + v) * 32 + 16 + ncol] = (_Float16)t1;
        }
        __syncthreads();
        v16h ha = load_a_frag(scL, 32, 0, lane);
#pragma unroll
        for (int nt = 0; nt < 16; ++nt) {
            v16h b = load_b_frag(wB2 + (size_t)nt * 16 * HID, HID, hb, lane);
            C2[nt] = wmma16(ha, b, C2[nt]);
        }
        __syncthreads();
    }
#pragma unroll
    for (int nt = 0; nt < 16; ++nt) {
        float bias = be2[nt * 16 + ncol];
#pragma unroll
        for (int v = 0; v < 8; ++v) {
            int m = mbase + v, e = base + m;
            if (e < Nedges)
                outEF[(size_t)e * 256 + nt * 16 + ncol] = (C2[nt][v] + bias) * dwL[m];
        }
    }

    // ------ q/e projections -> X[128x64] (row = edge*8+head, col = channel) ------
#pragma unroll 1
    for (int pe = 0; pe < 2; ++pe) {
        const _Float16* W  = pe ? wWpe : wWq;
        const float* bias  = pe ? bpe  : bq;
        int efoff = pe ? 256 : 0;             // query cols 0..255, edge cols 256..511
#pragma unroll 1
        for (int nt = 0; nt < 16; ++nt) {
            v8f c = (v8f){};
#pragma unroll
            for (int kt = 0; kt < 8; ++kt) {
                v16h a = load_a_frag(efL, EFW, efoff + kt * 32, lane);
                v16h b = load_b_frag(W + (size_t)nt * 16 * 256, 256, kt * 32, lane);
                c = wmma16(a, b, c);
            }
            float bs = bias[nt * 16 + ncol];
            int colg = nt * 16 + ncol;
            int cch  = (colg >> 3) + pe * 32; // channel index (q:0..31, e:32..63)
            int hh   = colg & 7;              // head
#pragma unroll
            for (int v = 0; v < 8; ++v)
                xL[((mbase + v) * 8 + hh) * 64 + cch] = (_Float16)(c[v] + bs);
        }
    }
    __syncthreads();

    // ---------------- attention MLP + softmax, per 16-row tile of X ----------------
#pragma unroll 1
    for (int rt = 0; rt < 8; ++rt) {
        const _Float16* xb = xL + rt * 16 * 64;
        v16h xa0 = load_a_frag(xb, 64, 0, lane);
        v16h xa1 = load_a_frag(xb, 64, 32, lane);
#pragma unroll
        for (int ot = 0; ot < 4; ++ot) {      // hidden = relu(X@A1 + a1) -> scL[16x64]
            v8f c = (v8f){};
            c = wmma16(xa0, load_b_frag(wA1 + (size_t)ot * 16 * 64, 64, 0,  lane), c);
            c = wmma16(xa1, load_b_frag(wA1 + (size_t)ot * 16 * 64, 64, 32, lane), c);
            float bs = a1b[ot * 16 + ncol];
#pragma unroll
            for (int v = 0; v < 8; ++v) {
                float t = c[v] + bs;
                scL[(mbase + v) * 64 + ot * 16 + ncol] = (_Float16)(t > 0.f ? t : 0.f);
            }
        }
        __syncthreads();
        v16h sa0 = load_a_frag(scL, 64, 0, lane);
        v16h sa1 = load_a_frag(scL, 64, 32, lane);
        v8f L0 = (v8f){}, L1 = (v8f){};
        L0 = wmma16(sa0, load_b_frag(wA2, 64, 0,  lane), L0);
        L0 = wmma16(sa1, load_b_frag(wA2, 64, 32, lane), L0);
        L1 = wmma16(sa0, load_b_frag(wA2 + 16 * 64, 64, 0,  lane), L1);
        L1 = wmma16(sa1, load_b_frag(wA2 + 16 * 64, 64, 32, lane), L1);
        float b0 = a2b[ncol], b1 = a2b[16 + ncol];
#pragma unroll
        for (int v = 0; v < 8; ++v) {
            float x0 = L0[v] + b0;
            float x1 = L1[v] + b1;
            float mx = fmaxf(x0, x1);         // softmax over 32 channels (16 lanes x 2)
#pragma unroll
            for (int sh = 8; sh >= 1; sh >>= 1) mx = fmaxf(mx, __shfl_xor(mx, sh, 32));
            float p0 = __expf(x0 - mx), p1 = __expf(x1 - mx);
            float sum = p0 + p1;
#pragma unroll
            for (int sh = 8; sh >= 1; sh >>= 1) sum += __shfl_xor(sum, sh, 32);
            float inv = 1.0f / sum;
            p0 *= inv; p1 *= inv;
            int R  = rt * 16 + mbase + v;     // X row = edge*8 + head
            int me = R >> 3, hh = R & 7, e = base + me;
            if (e < Nedges) {
                outP[(size_t)e * 256 + ncol * 8 + hh]        = p0;
                outP[(size_t)e * 256 + (16 + ncol) * 8 + hh] = p1;
            }
        }
        __syncthreads();
    }

    // ---------------- value_proj, scaled by dw/(dw+eps) -> x ----------------
#pragma unroll 1
    for (int nt = 0; nt < 16; ++nt) {
        v8f c = (v8f){};
#pragma unroll
        for (int kt = 0; kt < 8; ++kt) {
            v16h a = load_a_frag(efL, EFW, 768 + kt * 32, lane);   // value cols
            v16h b = load_b_frag(wWv + (size_t)nt * 16 * 256, 256, kt * 32, lane);
            c = wmma16(a, b, c);
        }
        float bias = bvv[nt * 16 + ncol];
#pragma unroll
        for (int v = 0; v < 8; ++v) {
            int m = mbase + v, e = base + m;
            if (e < Nedges)
                outX[(size_t)e * 256 + nt * 16 + ncol] = (c[v] + bias) * sL[m];
        }
    }
}

extern "C" void kernel_launch(void* const* d_in, const int* in_sizes, int n_in,
                              void* d_out, int out_size, void* d_ws, size_t ws_size,
                              hipStream_t stream) {
    const float* query = (const float*)d_in[0];
    const float* edge  = (const float*)d_in[1];
    const float* value = (const float*)d_in[2];
    const float* redge = (const float*)d_in[3];
    const float* srcc  = (const float*)d_in[4];
    const float* dstc  = (const float*)d_in[5];
    const float* We1 = (const float*)d_in[6];
    const float* be1 = (const float*)d_in[7];
    const float* We2 = (const float*)d_in[8];
    const float* be2 = (const float*)d_in[9];
    const float* Wq  = (const float*)d_in[10];
    const float* bq  = (const float*)d_in[11];
    const float* Wpe = (const float*)d_in[12];
    const float* bpe = (const float*)d_in[13];
    const float* Wv  = (const float*)d_in[14];
    const float* bv  = (const float*)d_in[15];
    const float* A1  = (const float*)d_in[16];
    const float* a1  = (const float*)d_in[17];
    const float* A2  = (const float*)d_in[18];
    const float* a2  = (const float*)d_in[19];
    int Nedges = in_sizes[0] / 256;

    _Float16* ws  = (_Float16*)d_ws;
    size_t o = 0;
    _Float16* wB1  = ws + o; o += (size_t)768 * 1024;
    _Float16* wB2  = ws + o; o += (size_t)256 * 768;
    _Float16* wWq  = ws + o; o += (size_t)256 * 256;
    _Float16* wWpe = ws + o; o += (size_t)256 * 256;
    _Float16* wWv  = ws + o; o += (size_t)256 * 256;
    _Float16* wA1  = ws + o; o += (size_t)64 * 64;
    _Float16* wA2  = ws + o; o += (size_t)32 * 64;

    auto cvt = [&](const float* W, _Float16* out, int K, int Nn) {
        int tot = K * Nn;
        cvtT_kernel<<<(tot + 255) / 256, 256, 0, stream>>>(W, out, K, Nn);
    };
    cvt(We1, wB1, 1024, 768);
    cvt(We2, wB2, 768, 256);
    cvt(Wq,  wWq, 256, 256);
    cvt(Wpe, wWpe, 256, 256);
    cvt(Wv,  wWv, 256, 256);
    cvt(A1,  wA1, 64, 64);
    cvt(A2,  wA2, 64, 32);

    float* outX  = (float*)d_out;
    float* outEF = outX  + (size_t)Nedges * 256;
    float* outP  = outEF + (size_t)Nedges * 256;

    int blocks = (Nedges + 15) / 16;
    mhea_kernel<<<blocks, 32, 0, stream>>>(query, edge, value, redge, srcc, dstc,
        be1, be2, bq, bpe, bv, a1, a2,
        wB1, wB2, wWq, wWpe, wWv, wA1, wA2,
        outX, outEF, outP, Nedges);
}
